// Head_15444702397231
// MI455X (gfx1250) — compile-verified, matrix operand order checked by reading, not run on hardware
//
#include <hip/hip_runtime.h>
#include <hip/hip_bf16.h>

// Single-head causal attention, B=1024 T=256 D=384 H=64.
// bf16 WMMA (v_wmma_f32_16x16x32_bf16) everywhere, f32 accumulate, f32 softmax.
// HBM floor ~0.7GB (~30us @ 23.3TB/s); bf16 WMMA keeps compute well below that.
// 32-row M-tiles in k_qkv (2 WMMAs per weight fragment); K/V staged into LDS
// in k_attn via GLOBAL_LOAD_ASYNC_TO_LDS_B128 (ASYNCcnt + s_wait_asynccnt).

typedef __bf16 bf16;
typedef __attribute__((ext_vector_type(16))) __bf16 v16bf;
typedef __attribute__((ext_vector_type(8)))  __bf16 v8bf;
typedef __attribute__((ext_vector_type(8)))  float  v8f;
typedef int v4i __attribute__((vector_size(16)));   // matches async builtin param

#define B_ 1024
#define T_ 256
#define D_ 384
#define H_ 64
#define XS_STRIDE 392   // 384+8 bf16 pad -> row stride 196 dw, %64=4: banks spread
#define KS_STRIDE 72    // 64+8 bf16 pad  -> 36 dw: banks spread
#define VS_STRIDE 264   // 256+8 bf16 pad -> 132 dw, %64=4: banks spread

#define SMEM_K_BYTES (T_ * KS_STRIDE * 2)             // 36864
#define SMEM_V_BYTES (H_ * VS_STRIDE * 2)             // 33792
#define SMEM_S_OFF   (SMEM_K_BYTES + SMEM_V_BYTES)    // 70656 (16B aligned)
#define SMEM_BYTES   (SMEM_S_OFF + 4 * 16 * 256 * 4)  // 136192

#if __has_builtin(__builtin_amdgcn_global_load_async_to_lds_b128)
#define HAVE_ASYNC_LDS 1
#else
#define HAVE_ASYNC_LDS 0
#endif

typedef __attribute__((address_space(1))) v4i* gv4i_p;  // global v4i*
typedef __attribute__((address_space(3))) v4i* lv4i_p;  // LDS v4i*

// 16B global->LDS copy: async (ASYNCcnt) when available, else sync load+store.
static __device__ __forceinline__ void cp16(const void* gsrc, void* ldst) {
#if HAVE_ASYNC_LDS
  __builtin_amdgcn_global_load_async_to_lds_b128(
      (gv4i_p)(void*)gsrc, (lv4i_p)ldst, 0, 0);
#else
  *(v4i*)ldst = *(const v4i*)gsrc;
#endif
}

static __device__ __forceinline__ void wait_async0() {
#if HAVE_ASYNC_LDS
#if __has_builtin(__builtin_amdgcn_s_wait_asynccnt)
  __builtin_amdgcn_s_wait_asynccnt(0);
#else
  asm volatile("s_wait_asynccnt 0" ::: "memory");
#endif
#endif
}

// Build v16bf A/B fragment from two contiguous 8-bf16 (16B) chunks.
// ISA 16-bit A 16x32 layout: lane (half=L>>4, m=L&15) holds
//   elems 0..7  = row m, K = k0 + 8*half + (0..7)
//   elems 8..15 = row m, K = k0 + 16 + 8*half + (0..7)
// B 32x16 mirrors with n = L&15 (B-frag == A-style load of the transpose).
static __device__ __forceinline__ v16bf frag16(const bf16* lo, const bf16* hi) {
  v8bf l = *(const v8bf*)lo;
  v8bf h = *(const v8bf*)hi;
  v16bf r;
#pragma unroll
  for (int i = 0; i < 8; ++i) { r[i] = l[i]; r[i + 8] = h[i]; }
  return r;
}

static __device__ __forceinline__ v8f wmma_bf16(v16bf a, v16bf b, v8f c) {
  return __builtin_amdgcn_wmma_f32_16x16x32_bf16(false, a, false, b, (short)0, c, false, false);
}

// ---------------------------------------------------------------------------
// Kernel 1: W [384,64] f32 -> WT [64,384] bf16 (convert + transpose).
// ---------------------------------------------------------------------------
__global__ __launch_bounds__(256) void k_wt(const float* __restrict__ W,
                                            bf16* __restrict__ WT) {
  for (int e = blockIdx.x * blockDim.x + threadIdx.x; e < D_ * H_;
       e += gridDim.x * blockDim.x) {
    int k = e / H_, n = e % H_;
    WT[n * D_ + k] = (bf16)W[e];
  }
}

// ---------------------------------------------------------------------------
// Kernel 2: QKV projection. Block = 32 rows of x (two M-tiles); 4 waves, wave
// w owns N-tile w for both M-tiles and all 3 matrices -> every weight B-frag
// feeds two WMMAs. x staged in LDS as bf16. Q,K row-major bf16 [B*T,64];
// V stored transposed Vt[b][h][t] bf16 for contiguous P@V B-fragments.
// ---------------------------------------------------------------------------
__global__ __launch_bounds__(128) void k_qkv(const float* __restrict__ x,
                                             const bf16* __restrict__ WqT,
                                             const bf16* __restrict__ WkT,
                                             const bf16* __restrict__ WvT,
                                             bf16* __restrict__ Q,
                                             bf16* __restrict__ K,
                                             bf16* __restrict__ Vt) {
  __shared__ __align__(16) bf16 xs[32 * XS_STRIDE];  // ~25 KB
  const int m0 = blockIdx.x * 32;

  for (int e = threadIdx.x; e < 32 * D_; e += 128) {
    int r = e / D_, c = e - r * D_;
    xs[r * XS_STRIDE + c] = (bf16)x[(size_t)(m0 + r) * D_ + c];
  }
  __syncthreads();

  const int lane = threadIdx.x & 31;
  const int w    = threadIdx.x >> 5;
  const int half = lane >> 4;
  const int lm   = lane & 15;
  const int n0   = w * 16;

  v8f cq0 = {}, ck0 = {}, cv0 = {};
  v8f cq1 = {}, ck1 = {}, cv1 = {};
  const bf16* wq = WqT + (size_t)(n0 + lm) * D_;
  const bf16* wk = WkT + (size_t)(n0 + lm) * D_;
  const bf16* wv = WvT + (size_t)(n0 + lm) * D_;
  const bf16* x0 = xs + lm * XS_STRIDE;
  const bf16* x1 = xs + (16 + lm) * XS_STRIDE;

#pragma unroll 2
  for (int k0 = 0; k0 < D_; k0 += 32) {
    const int o0 = k0 + 8 * half, o1 = k0 + 16 + 8 * half;
    v16bf a0 = frag16(x0 + o0, x0 + o1);
    v16bf a1 = frag16(x1 + o0, x1 + o1);
    v16bf bq = frag16(wq + o0, wq + o1);
    cq0 = wmma_bf16(a0, bq, cq0);
    cq1 = wmma_bf16(a1, bq, cq1);
    v16bf bk = frag16(wk + o0, wk + o1);
    ck0 = wmma_bf16(a0, bk, ck0);
    ck1 = wmma_bf16(a1, bk, ck1);
    v16bf bv = frag16(wv + o0, wv + o1);
    cv0 = wmma_bf16(a0, bv, cv0);
    cv1 = wmma_bf16(a1, bv, cv1);
  }

  const int b  = m0 / T_;
  const int t0 = m0 - b * T_;
  bf16* vtb = Vt + (size_t)b * H_ * T_ + (size_t)(n0 + lm) * T_;
#pragma unroll
  for (int r = 0; r < 8; ++r) {
    const int m = r + 8 * half;  // C/D: VGPR r, half-lane M offset 8
    size_t row0 = (size_t)(m0 + m);
    size_t row1 = (size_t)(m0 + 16 + m);
    Q[row0 * H_ + n0 + lm] = (bf16)cq0[r];
    Q[row1 * H_ + n0 + lm] = (bf16)cq1[r];
    K[row0 * H_ + n0 + lm] = (bf16)ck0[r];
    K[row1 * H_ + n0 + lm] = (bf16)ck1[r];
    vtb[t0 + m]      = (bf16)cv0[r];
    vtb[t0 + 16 + m] = (bf16)cv1[r];
  }
}

// ---------------------------------------------------------------------------
// Kernel 3: causal attention. Block = (batch, 64 q-rows); wave w owns q-tile
// qt = chunk*4 + w. K rows and Vt rows staged once per block into LDS via
// async global->LDS copies; S row tile (f32 16x256/wave) in LDS for exact
// softmax. Only tiles j <= qt are computed (causal skip).
// ---------------------------------------------------------------------------
__global__ __launch_bounds__(128) void k_attn(const bf16* __restrict__ Q,
                                              const bf16* __restrict__ K,
                                              const bf16* __restrict__ Vt,
                                              float* __restrict__ out) {
  extern __shared__ __align__(16) char smem[];
  bf16*  Ks = (bf16*)smem;                       // [T_][KS_STRIDE]
  bf16*  Vs = (bf16*)(smem + SMEM_K_BYTES);      // [H_][VS_STRIDE]
  float* S  = (float*)(smem + SMEM_S_OFF);       // [4][16][256]

  const int b     = blockIdx.x >> 2;
  const int chunk = blockIdx.x & 3;
  const int lane  = threadIdx.x & 31;
  const int w     = threadIdx.x >> 5;
  const int half  = lane >> 4;
  const int lm    = lane & 15;
  const int qt    = chunk * 4 + w;      // q-tile index within batch (0..15)
  const int q0    = qt * 16;
  const int nvalid = (qt + 1) * 16;     // causal: columns [0, nvalid)
  const int srows  = chunk * 64 + 64;   // max s-extent needed by this block

  const bf16* Qb = Q  + (size_t)b * T_ * H_;
  const bf16* Kb = K  + (size_t)b * T_ * H_;
  const bf16* Vb = Vt + (size_t)b * H_ * T_;
  float* Sw = S + w * (16 * 256);

  // Stage K rows [0, srows) : srows*8 16B chunks.
  for (int f = threadIdx.x; f < srows * 8; f += 128) {
    int r = f >> 3, c = (f & 7) * 8;
    cp16(Kb + (size_t)r * H_ + c, Ks + r * KS_STRIDE + c);
  }
  // Stage Vt rows (all 64 h), columns [0, srows) : 64 * srows/8 chunks.
  const int sc8 = srows >> 3;           // 16B chunks per Vt row
  for (int f = threadIdx.x; f < 64 * sc8; f += 128) {
    int hr = f / sc8, c = (f - hr * sc8) * 8;
    cp16(Vb + (size_t)hr * T_ + c, Vs + hr * VS_STRIDE + c);
  }

  // Q A-fragments (H=64 -> two K=32 fragments); each 128B row = 1 cacheline.
  const bf16* qrow = Qb + (size_t)(q0 + lm) * H_;
  v16bf qa0 = frag16(qrow + 8 * half,      qrow + 16 + 8 * half);
  v16bf qa1 = frag16(qrow + 32 + 8 * half, qrow + 48 + 8 * half);

  wait_async0();       // this wave's async copies done
  __syncthreads();     // all waves' copies visible

  // S = (Q K^T)/sqrt(H); B-frag of K^T == row load of staged K.
  for (int j = 0; j <= qt; ++j) {
    const int s0 = j * 16;
    const bf16* krow = Ks + (s0 + lm) * KS_STRIDE;
    v16bf kb0 = frag16(krow + 8 * half,      krow + 16 + 8 * half);
    v16bf kb1 = frag16(krow + 32 + 8 * half, krow + 48 + 8 * half);
    v8f c = {};
    c = wmma_bf16(qa0, kb0, c);
    c = wmma_bf16(qa1, kb1, c);
#pragma unroll
    for (int r = 0; r < 8; ++r) {
      const int m = r + 8 * half;
      float v = c[r] * 0.125f;                       // 1/sqrt(64)
      if (j == qt && lm > m) v = -__builtin_inff();  // causal mask on diagonal
      Sw[m * 256 + s0 + lm] = v;
    }
  }
  // Zero-pad one extra 16-col tile when nvalid % 32 != 0 so the K=32-step
  // P@V loop multiplies by zeros there.
  if ((qt & 1) == 0) {
    for (int idx = lane; idx < 256; idx += 32) {
      Sw[(idx >> 4) * 256 + nvalid + (idx & 15)] = 0.0f;
    }
  }

  // Row softmax (f32, wave-private).
  for (int m = 0; m < 16; ++m) {
    float* row = Sw + m * 256;
    float mx = -__builtin_inff();
    for (int c = lane; c < nvalid; c += 32) mx = fmaxf(mx, row[c]);
#pragma unroll
    for (int s = 16; s >= 1; s >>= 1) mx = fmaxf(mx, __shfl_xor(mx, s, 32));
    float sum = 0.0f;
    for (int c = lane; c < nvalid; c += 32) {
      float e = __expf(row[c] - mx);
      row[c] = e;
      sum += e;
    }
#pragma unroll
    for (int s = 16; s >= 1; s >>= 1) sum += __shfl_xor(sum, s, 32);
    const float inv = 1.0f / sum;
    for (int c = lane; c < nvalid; c += 32) row[c] *= inv;
  }

  // out = P @ V; A = P (LDS f32 -> bf16), B-frag of V == row load of staged Vt.
  v8f o[4] = {};
  const float* prow = Sw + lm * 256;
  for (int k0 = 0; k0 < nvalid; k0 += 32) {
    v16bf pa;
#pragma unroll
    for (int i = 0; i < 8; ++i) {
      pa[i]     = (bf16)prow[k0 + 8 * half + i];
      pa[i + 8] = (bf16)prow[k0 + 16 + 8 * half + i];
    }
#pragma unroll
    for (int nt = 0; nt < 4; ++nt) {
      const bf16* vrow = Vs + (nt * 16 + lm) * VS_STRIDE;
      v16bf vb = frag16(vrow + k0 + 8 * half, vrow + k0 + 16 + 8 * half);
      o[nt] = wmma_bf16(pa, vb, o[nt]);
    }
  }

#pragma unroll
  for (int nt = 0; nt < 4; ++nt) {
#pragma unroll
    for (int r = 0; r < 8; ++r) {
      const int m = r + 8 * half;
      out[((size_t)b * T_ + q0 + m) * H_ + nt * 16 + lm] = o[nt][r];
    }
  }
}

// ---------------------------------------------------------------------------
extern "C" void kernel_launch(void* const* d_in, const int* in_sizes, int n_in,
                              void* d_out, int out_size, void* d_ws, size_t ws_size,
                              hipStream_t stream) {
  (void)in_sizes; (void)n_in; (void)out_size; (void)ws_size;
  const float* x  = (const float*)d_in[0];
  const float* Wq = (const float*)d_in[1];
  const float* Wk = (const float*)d_in[2];
  const float* Wv = (const float*)d_in[3];

  char* ws = (char*)d_ws;
  const size_t wt_bytes = (size_t)H_ * D_ * sizeof(bf16);        // 48 KB each
  const size_t qk_bytes = (size_t)B_ * T_ * H_ * sizeof(bf16);   // 32 MiB each
  bf16* WqT = (bf16*)(ws);
  bf16* WkT = (bf16*)(ws + wt_bytes);
  bf16* WvT = (bf16*)(ws + 2 * wt_bytes);
  bf16* Qd  = (bf16*)(ws + 3 * wt_bytes);
  bf16* Kd  = (bf16*)(ws + 3 * wt_bytes + qk_bytes);
  bf16* Vtd = (bf16*)(ws + 3 * wt_bytes + 2 * qk_bytes);
  // total ws use ~96.2 MB

  // Allow >64KB dynamic LDS for k_attn (host-side setting; not a stream op).
  (void)hipFuncSetAttribute((const void*)k_attn,
                            hipFuncAttributeMaxDynamicSharedMemorySize,
                            SMEM_BYTES);

  k_wt  <<<24, 256, 0, stream>>>(Wq, WqT);
  k_wt  <<<24, 256, 0, stream>>>(Wk, WkT);
  k_wt  <<<24, 256, 0, stream>>>(Wv, WvT);
  k_qkv <<<(B_ * T_) / 32, 128, 0, stream>>>(x, WqT, WkT, WvT, Qd, Kd, Vtd);
  k_attn<<<B_ * 4, 128, SMEM_BYTES, stream>>>(Qd, Kd, Vtd, (float*)d_out);
}
